// MultiHeadSelfAttention_17772574671217
// MI455X (gfx1250) — matrix operand order repaired and checked by reading, not compiled, not run on hardware
//
#include <hip/hip_runtime.h>

// ---------------------------------------------------------------------------
// MI455X (gfx1250) multi-head self-attention, bf16 WMMA pipeline.
// B=8, N=1024, D=768, H=12, HD=64.  All matmuls via v_wmma_f32_16x16x32_bf16.
// K/V^T chunks staged into LDS with global_load_async_to_lds_b128 (ASYNCcnt).
// ---------------------------------------------------------------------------

typedef __attribute__((ext_vector_type(16))) __bf16 v16bf;
typedef __attribute__((ext_vector_type(8)))  float  v8f;

union FragBF {
    v16bf v;
    unsigned int w[8];
    unsigned short u[16];
};

// f32 -> bf16 (RNE) through the native conversion path: clang lowers the
// __bf16 cast to the hardware convert on gfx1250 instead of a 4-op bit hack.
__device__ __forceinline__ unsigned short f2bf(float f) {
    union { __bf16 b; unsigned short u; } cv;
    cv.b = (__bf16)f;
    return cv.u;
}

#if __has_builtin(__builtin_amdgcn_cvt_pk_bf16_f32)
__device__ __forceinline__ unsigned int pack2bf(float a, float b) {
    auto r = __builtin_amdgcn_cvt_pk_bf16_f32(a, b);   // packed native convert
    unsigned int u;
    __builtin_memcpy(&u, &r, sizeof(u));
    return u;
}
#else
__device__ __forceinline__ unsigned int pack2bf(float a, float b) {
    return (unsigned int)f2bf(a) | ((unsigned int)f2bf(b) << 16);
}
#endif

// A-operand (16x32 bf16) k index for VGPR v, lane-half lh:  row = lane%16
__device__ __forceinline__ int kA(int v, int lh) {
    return (v >> 2) * 16 + lh * 8 + (v & 3) * 2;
}
// B-operand (32x16 bf16) k index for VGPR v, lane-half lh:  col = lane%16
__device__ __forceinline__ int kB(int v, int lh) {
    return (v >> 2) * 16 + lh * 8 + (v & 3) * 2;
}

#define WMMA_BF16(A, B, C) \
    __builtin_amdgcn_wmma_f32_16x16x32_bf16(false, (A), false, (B), (short)0, (C), false, false)

// Flat shared-memory address -> 32-bit LDS byte offset (ISA: LDS aperture
// truncates to addr[31:0]).
__device__ __forceinline__ unsigned lds_off(const void* p) {
    return (unsigned)(unsigned long long)p;
}
// Async DMA 16B/lane from global to LDS, tracked by ASYNCcnt.
__device__ __forceinline__ void async_copy_b128(unsigned dst_lds, const void* src) {
    asm volatile("global_load_async_to_lds_b128 %0, %1, off"
                 :: "v"(dst_lds), "v"(src) : "memory");
}
__device__ __forceinline__ void async_wait0() {
    asm volatile("s_wait_asynccnt 0" ::: "memory");
}

// ---------------------------------------------------------------------------
// Kernel 1: per-head QKV projection.  q = x_h @ Wq + bq  (and K, V).
// Block = (head h, batch b, 128-row chunk), 8 waves, each wave a 16-row tile.
// Writes Q,K row-major bf16 [B,H,N,64];  V transposed bf16 [B,H,64,N].
// ---------------------------------------------------------------------------
__global__ void __launch_bounds__(256) qkv_kernel(
    const float* __restrict__ x,
    const float* __restrict__ Wq, const float* __restrict__ Wk, const float* __restrict__ Wv,
    const float* __restrict__ bq, const float* __restrict__ bk, const float* __restrict__ bv,
    unsigned short* __restrict__ Qb, unsigned short* __restrict__ Kb,
    unsigned short* __restrict__ VTb)
{
    const int h = blockIdx.x, b = blockIdx.y, nchunk = blockIdx.z;
    const int t = threadIdx.x, wave = t >> 5, lane = t & 31;
    const int lm = lane & 15, lh = lane >> 4;

    __shared__ __align__(16) unsigned short Wt[3][64 * 64];   // [mat][e][d] = W[h][d][e]

#pragma unroll
    for (int mat = 0; mat < 3; ++mat) {
        const float* W = (mat == 0 ? Wq : mat == 1 ? Wk : Wv) + h * 64 * 64;
#pragma unroll
        for (int i = 0; i < 16; ++i) {
            int lin = t + 256 * i;
            int d = lin >> 6, e = lin & 63;
            Wt[mat][e * 64 + d] = f2bf(W[d * 64 + e]);
        }
    }
    __syncthreads();

    const int n0 = nchunk * 128 + wave * 16;
    const long xrow = ((long)b * 1024 + n0 + lm) * 768 + h * 64;

    FragBF a[2];
#pragma unroll
    for (int g = 0; g < 2; ++g)
#pragma unroll
        for (int v = 0; v < 8; ++v) {
            int k = g * 32 + kA(v, lh);
            float2 f = *(const float2*)(x + xrow + k);
            a[g].w[v] = pack2bf(f.x, f.y);
        }

#pragma unroll
    for (int mat = 0; mat < 3; ++mat) {
        const float* bias = (mat == 0 ? bq : mat == 1 ? bk : bv) + h * 64;
#pragma unroll
        for (int nt = 0; nt < 4; ++nt) {
            v8f c = {};
#pragma unroll
            for (int g = 0; g < 2; ++g) {
                FragBF bf;
                int e = nt * 16 + lm;
#pragma unroll
                for (int v = 0; v < 8; ++v)
                    bf.w[v] = *(const unsigned int*)&Wt[mat][e * 64 + g * 32 + kB(v, lh)];
                c = WMMA_BF16(a[g].v, bf.v, c);
            }
            const float be = bias[nt * 16 + lm];
            if (mat < 2) {
                unsigned short* dst = (mat == 0 ? Qb : Kb) + (long)(b * 12 + h) * 1024 * 64;
#pragma unroll
                for (int v = 0; v < 8; ++v) {
                    int n = n0 + v + 8 * lh;
                    dst[(long)n * 64 + nt * 16 + lm] = f2bf(c[v] + be);
                }
            } else {
                unsigned short* dst = VTb + (long)(b * 12 + h) * 64 * 1024;
#pragma unroll
                for (int v = 0; v < 8; ++v) {
                    int n = n0 + v + 8 * lh;
                    dst[(long)(nt * 16 + lm) * 1024 + n] = f2bf(c[v] + be);
                }
            }
        }
    }
}

// ---------------------------------------------------------------------------
// Kernel 2: flash attention per (b,h).  Each wave owns 16 query rows; block
// of 8 waves shares async-LDS-staged K / V^T chunks of 64 keys per iteration.
// Online softmax; scale = D^-0.5 (faithful reference quirk).
// ---------------------------------------------------------------------------
__global__ void __launch_bounds__(256) attn_kernel(
    const unsigned short* __restrict__ Qb, const unsigned short* __restrict__ Kb,
    const unsigned short* __restrict__ VTb, unsigned short* __restrict__ AOb)
{
    const int nchunk = blockIdx.x, h = blockIdx.y, b = blockIdx.z;
    const int t = threadIdx.x, wave = t >> 5, lane = t & 31;
    const int lm = lane & 15, lh = lane >> 4;

    __shared__ __align__(16) unsigned short Kl[64 * 64];      // [j][d]
    __shared__ __align__(16) unsigned short VTl[64 * 64];     // [e][j]
    __shared__ __align__(16) unsigned short Pl[8][16 * 64];   // per-wave P tile [m][j]

    const long bh = (long)(b * 12 + h);
    const unsigned short* Qp = Qb + bh * 1024 * 64;
    const unsigned short* Kp = Kb + bh * 1024 * 64;
    const unsigned short* Vp = VTb + bh * 64 * 1024;

    const int q0 = nchunk * 128 + wave * 16;

    FragBF qa[2];
#pragma unroll
    for (int g = 0; g < 2; ++g)
#pragma unroll
        for (int v = 0; v < 8; ++v)
            qa[g].w[v] = *(const unsigned int*)&Qp[(long)(q0 + lm) * 64 + g * 32 + kA(v, lh)];

    v8f o[4] = {};
    float mrun[8], lrun[8];
#pragma unroll
    for (int v = 0; v < 8; ++v) { mrun[v] = -1e30f; lrun[v] = 0.0f; }
    const float scale = 0.03608439182435161f;   // 768^-0.5

    for (int j0 = 0; j0 < 1024; j0 += 64) {
        // Async DMA stage: K chunk 8KB (contiguous) + V^T chunk 8KB.
#pragma unroll
        for (int i = 0; i < 2; ++i) {
            int idx = t + 256 * i;
            async_copy_b128(lds_off(&Kl[idx * 8]), Kp + (long)j0 * 64 + idx * 8);
            int e = idx >> 3, p = idx & 7;
            async_copy_b128(lds_off(&VTl[e * 64 + p * 8]), Vp + (long)e * 1024 + j0 + p * 8);
        }
        async_wait0();
        __syncthreads();

        // S = Q K^T : four 16-key subtiles, two k-steps each.
        v8f s[4];
#pragma unroll
        for (int jj = 0; jj < 4; ++jj) {
            v8f c = {};
#pragma unroll
            for (int g = 0; g < 2; ++g) {
                FragBF kb;
                int jrow = jj * 16 + lm;
#pragma unroll
                for (int v = 0; v < 8; ++v)
                    kb.w[v] = *(const unsigned int*)&Kl[jrow * 64 + g * 32 + kB(v, lh)];
                c = WMMA_BF16(qa[g].v, kb.v, c);
            }
            s[jj] = c;
        }

        // Online softmax over 64 keys; C-layout row = v + 8*lh, col = lm.
#pragma unroll
        for (int v = 0; v < 8; ++v) {
            float s0 = s[0][v] * scale, s1 = s[1][v] * scale;
            float s2 = s[2][v] * scale, s3 = s[3][v] * scale;
            float mx = fmaxf(fmaxf(s0, s1), fmaxf(s2, s3));
#pragma unroll
            for (int m = 1; m < 16; m <<= 1) mx = fmaxf(mx, __shfl_xor(mx, m, 32));
            float mnew  = fmaxf(mrun[v], mx);
            float alpha = __expf(mrun[v] - mnew);
            float p0 = __expf(s0 - mnew), p1 = __expf(s1 - mnew);
            float p2 = __expf(s2 - mnew), p3 = __expf(s3 - mnew);
            float rs = (p0 + p1) + (p2 + p3);
#pragma unroll
            for (int m = 1; m < 16; m <<= 1) rs += __shfl_xor(rs, m, 32);
            lrun[v] = lrun[v] * alpha + rs;
            mrun[v] = mnew;
#pragma unroll
            for (int nt = 0; nt < 4; ++nt) o[nt][v] *= alpha;
            int row = v + 8 * lh;
            Pl[wave][row * 64 +      lm] = f2bf(p0);
            Pl[wave][row * 64 + 16 + lm] = f2bf(p1);
            Pl[wave][row * 64 + 32 + lm] = f2bf(p2);
            Pl[wave][row * 64 + 48 + lm] = f2bf(p3);
        }

        // Re-read P in A-fragment layout (same-wave DS ops complete in order).
        FragBF pa[2];
#pragma unroll
        for (int g = 0; g < 2; ++g)
#pragma unroll
            for (int v = 0; v < 8; ++v)
                pa[g].w[v] = *(const unsigned int*)&Pl[wave][lm * 64 + g * 32 + kA(v, lh)];

        // O += P @ V   (B operand = rows of V^T from LDS).
#pragma unroll
        for (int nt = 0; nt < 4; ++nt) {
#pragma unroll
            for (int g = 0; g < 2; ++g) {
                FragBF vb;
                int e = nt * 16 + lm;
#pragma unroll
                for (int v = 0; v < 8; ++v)
                    vb.w[v] = *(const unsigned int*)&VTl[e * 64 + g * 32 + kB(v, lh)];
                o[nt] = WMMA_BF16(pa[g].v, vb.v, o[nt]);
            }
        }
        __syncthreads();
    }

    // Normalize and store concat-head layout [B,N,D] bf16.
#pragma unroll
    for (int nt = 0; nt < 4; ++nt)
#pragma unroll
        for (int v = 0; v < 8; ++v) {
            float val = o[nt][v] / lrun[v];
            int n = q0 + v + 8 * lh;
            AOb[((long)b * 1024 + n) * 768 + h * 64 + nt * 16 + lm] = f2bf(val);
        }
}

// ---------------------------------------------------------------------------
// Kernel 3: output projection  Y = AO @ Wo^T + bo.
// Block = (128-row chunk, 64-col chunk); Wo chunk staged bf16 in LDS per
// k-step (Wo rows are exactly B^T rows, so no transpose needed).
// ---------------------------------------------------------------------------
__global__ void __launch_bounds__(256) proj_kernel(
    const unsigned short* __restrict__ AOb, const float* __restrict__ Wo,
    const float* __restrict__ bo, float* __restrict__ out)
{
    const int rowchunk = blockIdx.x;
    const int o0 = blockIdx.y * 64;
    const int t = threadIdx.x, wave = t >> 5, lane = t & 31;
    const int lm = lane & 15, lh = lane >> 4;

    __shared__ __align__(16) unsigned short Wl[64 * 32];   // [o][d]

    const int row0 = rowchunk * 128 + wave * 16;
    v8f c[4] = {};

    for (int ks = 0; ks < 24; ++ks) {
        {   // stage Wo[o0:o0+64][ks*32 : ks*32+32] as bf16
            int oo = t >> 2, dp = (t & 3) * 8;
            const float* src = Wo + (long)(o0 + oo) * 768 + ks * 32 + dp;
            float4 f0 = *(const float4*)src;
            float4 f1 = *(const float4*)(src + 4);
            unsigned int wbuf[4] = { pack2bf(f0.x, f0.y), pack2bf(f0.z, f0.w),
                                     pack2bf(f1.x, f1.y), pack2bf(f1.z, f1.w) };
            *(uint4*)&Wl[oo * 32 + dp] = *(const uint4*)wbuf;
        }
        __syncthreads();

        FragBF a;
#pragma unroll
        for (int v = 0; v < 8; ++v)
            a.w[v] = *(const unsigned int*)&AOb[(long)(row0 + lm) * 768 + ks * 32 + kA(v, lh)];
#pragma unroll
        for (int nt = 0; nt < 4; ++nt) {
            FragBF bf;
            int oo = nt * 16 + lm;
#pragma unroll
            for (int v = 0; v < 8; ++v)
                bf.w[v] = *(const unsigned int*)&Wl[oo * 32 + kB(v, lh)];
            c[nt] = WMMA_BF16(a.v, bf.v, c[nt]);
        }
        __syncthreads();
    }

#pragma unroll
    for (int nt = 0; nt < 4; ++nt) {
        float be = bo[o0 + nt * 16 + lm];
#pragma unroll
        for (int v = 0; v < 8; ++v) {
            int r = row0 + v + 8 * lh;
            out[(long)r * 768 + o0 + nt * 16 + lm] = c[nt][v] + be;
        }
    }
}

// ---------------------------------------------------------------------------
extern "C" void kernel_launch(void* const* d_in, const int* in_sizes, int n_in,
                              void* d_out, int out_size, void* d_ws, size_t ws_size,
                              hipStream_t stream)
{
    (void)in_sizes; (void)n_in; (void)out_size; (void)ws_size;

    const float* x  = (const float*)d_in[0];
    const float* Wq = (const float*)d_in[1];
    const float* Wk = (const float*)d_in[2];
    const float* Wv = (const float*)d_in[3];
    const float* bq = (const float*)d_in[4];
    const float* bk = (const float*)d_in[5];
    const float* bv = (const float*)d_in[6];
    const float* Wo = (const float*)d_in[7];
    const float* bo = (const float*)d_in[8];

    char* ws = (char*)d_ws;
    const size_t nQ = (size_t)8 * 12 * 1024 * 64;   // == B*N*D element count
    unsigned short* Qb  = (unsigned short*)ws;  ws += nQ * 2;
    unsigned short* Kb  = (unsigned short*)ws;  ws += nQ * 2;
    unsigned short* VTb = (unsigned short*)ws;  ws += nQ * 2;
    unsigned short* AOb = (unsigned short*)ws;  ws += nQ * 2;

    qkv_kernel <<<dim3(12, 8, 8), 256, 0, stream>>>(x, Wq, Wk, Wv, bq, bk, bv, Qb, Kb, VTb);
    attn_kernel<<<dim3(8, 12, 8), 256, 0, stream>>>(Qb, Kb, VTb, AOb);
    proj_kernel<<<dim3(64, 12),   256, 0, stream>>>(AOb, Wo, bo, (float*)d_out);
}
